// HybridLayer_43224550867456
// MI455X (gfx1250) — matrix operand order repaired and checked by reading, not compiled, not run on hardware
//
#include <hip/hip_runtime.h>
#include <math.h>

typedef _Float16 half_t;
typedef __attribute__((ext_vector_type(16))) _Float16 v16h;
typedef __attribute__((ext_vector_type(8)))  float    v8f;

union F16x16 { v16h v; uint4 q[2]; };

#if __has_builtin(__builtin_amdgcn_global_load_async_to_lds_b128)
#define ASYNC_LDS 1
typedef int vint4_t __attribute__((ext_vector_type(4)));
typedef __attribute__((address_space(1))) vint4_t gvint4_t;  // global (AS1)
typedef __attribute__((address_space(3))) vint4_t lvint4_t;  // LDS (AS3)
#else
#define ASYNC_LDS 0
#endif

__device__ __forceinline__ void wait_async_lds()
{
#if ASYNC_LDS
#if __has_builtin(__builtin_amdgcn_s_wait_asynccnt)
    __builtin_amdgcn_s_wait_asynccnt(0);
#else
    asm volatile("s_wait_asynccnt 0" ::: "memory");
#endif
#endif
}

// ---------------------------------------------------------------------------
// WMMA GEMM: C[M,N] = A[M,K](f16, row-major) * B (pre-packed f16 pairs)
// B is pre-packed "fragment-ready": Bp[n*(K/2) + k/2] = {B[k][n], B[k+1][n]}.
// Requires M%64==0, N%BN==0, K%32==0.
// Block = 128 threads (4 waves, 2x2); block tile 64 x BN; wave tile 32 x BN/2.
// Both tiles staged with async global->LDS (ASYNCcnt), double-buffered:
// issue loads for tile k+1 after the barrier, compute tile k, wait+barrier.
// ---------------------------------------------------------------------------
template <int EPI, int BN> // EPI: 0 = store, 1 = store + residual add
__global__ __launch_bounds__(128)
void gemm_wmma_f16(const half_t* __restrict__ A, const unsigned* __restrict__ Bp,
                   float* __restrict__ C, const float* __restrict__ Res,
                   int M, int N, int K)
{
    constexpr int BM = 64, BK = 32;
    constexpr int JT = BN / 32;                    // B tiles per wave (2 or 4)
    __shared__ half_t   As[2][BM][BK + 8];         // stride 40 halfs (80B)
    __shared__ unsigned Bt[2][BN][BK / 2 + 4];     // stride 20 dwords (80B)

    const int tid     = threadIdx.x;
    const int lane    = tid & 31;
    const int wave    = tid >> 5;               // 0..3
    const int wm      = (wave >> 1) * 32;       // wave tile row
    const int wn      = (wave & 1) * (BN / 2);  // wave tile col
    const int m0      = blockIdx.y * BM;
    const int n0      = blockIdx.x * BN;
    const int halfSel = lane >> 4;
    const int l15     = lane & 15;
    const int Khalf   = K >> 1;

    v8f acc[2][JT] = {};

    // stage tile (k0) into LDS buffer b : pure b128 copies (async when avail)
    auto stage = [&](int k0, int b) {
        // A tile: 64x32 halfs = 256 x b128, 2 per thread
        #pragma unroll
        for (int v = tid; v < (BM * BK) / 8; v += 128) {
            int r  = v >> 2;
            int cv = v & 3;
            const half_t* gsrc = A + (size_t)(m0 + r) * K + k0 + cv * 8;
            half_t* ldst = &As[b][r][cv * 8];
#if ASYNC_LDS
            __builtin_amdgcn_global_load_async_to_lds_b128(
                (gvint4_t*)gsrc, (lvint4_t*)ldst, 0, 0);
#else
            *reinterpret_cast<uint4*>(ldst) =
                *reinterpret_cast<const uint4*>(gsrc);
#endif
        }
        // B tile (pre-packed): BN rows x 16 dwords = BN*4 x b128
        #pragma unroll
        for (int v = tid; v < BN * 4; v += 128) {
            int r  = v >> 2;
            int cv = v & 3;
            const unsigned* gsrc = Bp + (size_t)(n0 + r) * Khalf + (k0 >> 1) + cv * 4;
            unsigned* ldst = &Bt[b][r][cv * 4];
#if ASYNC_LDS
            __builtin_amdgcn_global_load_async_to_lds_b128(
                (gvint4_t*)gsrc, (lvint4_t*)ldst, 0, 0);
#else
            *reinterpret_cast<uint4*>(ldst) =
                *reinterpret_cast<const uint4*>(gsrc);
#endif
        }
    };

    stage(0, 0);
    int buf = 0;
    for (int k0 = 0; k0 < K; k0 += BK, buf ^= 1) {
        wait_async_lds();       // my wave's loads for tile k done
        __syncthreads();        // everyone's loads visible; prior reads done
        if (k0 + BK < K) stage(k0 + BK, buf ^ 1);   // overlap with compute

        // ---- A fragments: 16-bit A 16x32 layout, 2 x b128 per tile ----
        v16h afrag[2], bfrag[JT];
        #pragma unroll
        for (int i = 0; i < 2; ++i) {
            const half_t* base = &As[buf][wm + i * 16 + l15][halfSel * 8];
            F16x16 t;
            t.q[0] = *reinterpret_cast<const uint4*>(base);       // K = 8h..8h+7
            t.q[1] = *reinterpret_cast<const uint4*>(base + 16);  // K = 8h+16..8h+23
            afrag[i] = t.v;
        }
        // ---- B fragments: VGPR v = K pair (8h+2v) for v<4, (16+8h+2(v-4)) for v>=4, 2 x b128 per tile ----
        #pragma unroll
        for (int j = 0; j < JT; ++j) {
            const unsigned* bp = &Bt[buf][wn + j * 16 + l15][halfSel * 4];
            F16x16 t;
            t.q[0] = *reinterpret_cast<const uint4*>(bp);
            t.q[1] = *reinterpret_cast<const uint4*>(bp + 8);
            bfrag[j] = t.v;
        }

        #pragma unroll
        for (int i = 0; i < 2; ++i)
            #pragma unroll
            for (int j = 0; j < JT; ++j)
                acc[i][j] = __builtin_amdgcn_wmma_f32_16x16x32_f16(
                    false, afrag[i], false, bfrag[j],
                    (short)0, acc[i][j], false, false);
    }

    // ---- epilogue: C/D layout row = v + 8*halfSel, col = lane&15 ----
    #pragma unroll
    for (int i = 0; i < 2; ++i) {
        #pragma unroll
        for (int j = 0; j < JT; ++j) {
            #pragma unroll
            for (int v = 0; v < 8; ++v) {
                int row = m0 + wm + i * 16 + v + 8 * halfSel;
                int col = n0 + wn + j * 16 + l15;
                float val = acc[i][j][v];
                size_t off = (size_t)row * N + col;
                if (EPI == 1) val += Res[off];
                C[off] = val;
            }
        }
    }
}

// ---------------------------------------------------------------------------
// weight pack: W[K][N] fp32 (row-major) -> Wp[n][k/2] packed f16-pair dwords
// Wp[n*(K/2)+p] = {f16(W[2p][n]) , f16(W[2p+1][n])<<16}
// ---------------------------------------------------------------------------
__global__ void pack_weight_f16(const float* __restrict__ W, unsigned* __restrict__ Wp,
                                int K, int N)
{
    int idx = blockIdx.x * 256 + threadIdx.x;
    int total = (K >> 1) * N;
    if (idx >= total) return;
    int n = idx % N;          // consecutive threads -> coalesced reads
    int p = idx / N;
    half_t lo = (half_t)W[(size_t)(2 * p) * N + n];
    half_t hi = (half_t)W[(size_t)(2 * p + 1) * N + n];
    unsigned short ulo, uhi;
    __builtin_memcpy(&ulo, &lo, 2);
    __builtin_memcpy(&uhi, &hi, 2);
    Wp[(size_t)n * (K >> 1) + p] = (unsigned)ulo | ((unsigned)uhi << 16);
}

// ---------------------------------------------------------------------------
// RMSNorm (fp32 in) -> f16 out, one block per row
// ---------------------------------------------------------------------------
__global__ __launch_bounds__(256)
void rmsnorm_f16(const float* __restrict__ x, const float* __restrict__ w,
                 half_t* __restrict__ out, int D)
{
    __shared__ float red[8];
    const int row = blockIdx.x;
    const float* xr = x + (size_t)row * D;
    float s = 0.f;
    for (int i = threadIdx.x; i < D; i += 256) { float v = xr[i]; s += v * v; }
    #pragma unroll
    for (int off = 16; off > 0; off >>= 1) s += __shfl_down(s, off);
    const int lane = threadIdx.x & 31, wv = threadIdx.x >> 5;
    if (lane == 0) red[wv] = s;
    __syncthreads();
    if (wv == 0) {
        float t = (lane < 8) ? red[lane] : 0.f;
        #pragma unroll
        for (int off = 4; off > 0; off >>= 1) t += __shfl_down(t, off);
        if (lane == 0) red[0] = t;
    }
    __syncthreads();
    const float scale = rsqrtf(red[0] / (float)D + 1e-5f);
    for (int i = threadIdx.x; i < D; i += 256)
        out[(size_t)row * D + i] = (half_t)(xr[i] * scale * w[i]);
}

// ---------------------------------------------------------------------------
// depthwise conv (K=4, SAME: t-1+j) + SiLU ; outputs fp32 + f16
// ---------------------------------------------------------------------------
__global__ void conv_silu(const float* __restrict__ xp, const float* __restrict__ cw,
                          const float* __restrict__ cb, float* __restrict__ xc32,
                          half_t* __restrict__ xc16, int T, int E, int NP)
{
    int idx = blockIdx.x * 256 + threadIdx.x;
    if (idx >= T * E) return;
    int t = idx / E, e = idx % E;
    float acc = cb[e];
    #pragma unroll
    for (int j = 0; j < 4; ++j) {
        int tt = t - 1 + j;
        if (tt >= 0 && tt < T) acc += xp[(size_t)tt * NP + e] * cw[j * E + e];
    }
    float s = acc / (1.f + __expf(-acc));
    xc32[idx] = s;
    xc16[idx] = (half_t)s;
}

// ---------------------------------------------------------------------------
// dt = clip(softplus(v + bias), 1e-3, 0.1) in place
// ---------------------------------------------------------------------------
__global__ void softplus_clip(float* __restrict__ v, const float* __restrict__ bias,
                              int n, int E)
{
    int i = blockIdx.x * 256 + threadIdx.x;
    if (i >= n) return;
    float x = v[i] + bias[i & (E - 1)];
    float sp = (x > 20.f) ? x : log1pf(__expf(x));
    v[i] = fminf(fmaxf(sp, 0.001f), 0.1f);
}

// ---------------------------------------------------------------------------
// selective-scan: one thread per channel e, h[64] in VGPRs, B/C chunks in LDS
// ---------------------------------------------------------------------------
__global__ __launch_bounds__(64)
void ssm_scan(const float* __restrict__ xc, const float* __restrict__ dt,
              const float* __restrict__ Bp, const float* __restrict__ Cp,
              const float* __restrict__ A_log, float* __restrict__ y,
              int T, int E)
{
    constexpr int NS = 64, TCHUNK = 16;
    __shared__ float Av[NS];
    __shared__ float Bsh[TCHUNK][NS];
    __shared__ float Csh[TCHUNK][NS];
    const int e = blockIdx.x * 64 + threadIdx.x;
    if (threadIdx.x < NS) Av[threadIdx.x] = -__expf(A_log[threadIdx.x]);

    float h[NS];
    #pragma unroll
    for (int n = 0; n < NS; ++n) h[n] = 0.f;

    for (int t0 = 0; t0 < T; t0 += TCHUNK) {
        __syncthreads();
        for (int i = threadIdx.x; i < TCHUNK * NS; i += 64) {
            int tt = i >> 6, nn = i & 63;
            Bsh[tt][nn] = Bp[(size_t)(t0 + tt) * NS + nn];
            Csh[tt][nn] = Cp[(size_t)(t0 + tt) * NS + nn];
        }
        __syncthreads();
        #pragma unroll 4
        for (int tt = 0; tt < TCHUNK; ++tt) {
            const int t  = t0 + tt;
            const float d  = dt[(size_t)t * E + e];
            const float dx = d * xc[(size_t)t * E + e];
            float acc = 0.f;
            #pragma unroll
            for (int n = 0; n < NS; ++n) {
                float hn = __expf(d * Av[n]) * h[n] + dx * Bsh[tt][n];
                h[n] = hn;
                acc += hn * Csh[tt][n];
            }
            y[(size_t)t * E + e] = acc;
        }
    }
}

// ---------------------------------------------------------------------------
// y *= silu(z); z lives in xp[:, E:2E] ; emit f16
// ---------------------------------------------------------------------------
__global__ void gate_mul(const float* __restrict__ xp, const float* __restrict__ y,
                         half_t* __restrict__ y16, int T, int E, int NP)
{
    int idx = blockIdx.x * 256 + threadIdx.x;
    if (idx >= T * E) return;
    int t = idx / E, e = idx % E;
    float z = xp[(size_t)t * NP + E + e];
    float g = z / (1.f + __expf(-z));
    y16[idx] = (half_t)(y[idx] * g);
}

// ---------------------------------------------------------------------------
// a = silu(g) * u -> f16
// ---------------------------------------------------------------------------
__global__ void silu_mul(const float* __restrict__ g, const float* __restrict__ u,
                         half_t* __restrict__ out, int n)
{
    int i = blockIdx.x * 256 + threadIdx.x;
    if (i >= n) return;
    float gv = g[i];
    float s = gv / (1.f + __expf(-gv));
    out[i] = (half_t)(s * u[i]);
}

// ---------------------------------------------------------------------------
// host-side launcher
// ---------------------------------------------------------------------------
static inline size_t align_up(size_t v) { return (v + 255) & ~size_t(255); }

extern "C" void kernel_launch(void* const* d_in, const int* in_sizes, int n_in,
                              void* d_out, int out_size, void* d_ws, size_t ws_size,
                              hipStream_t stream)
{
    (void)in_sizes; (void)n_in; (void)out_size; (void)ws_size;
    const int T = 2048, D = 1024, E = 2048, NS = 64, I = 2048;
    const int NP = 2 * E; // in_proj output width = 4096

    const float* x        = (const float*)d_in[0];
    const float* in_nw    = (const float*)d_in[1];
    const float* ffn_nw   = (const float*)d_in[2];
    const float* in_projW = (const float*)d_in[3];
    const float* convW    = (const float*)d_in[4];
    const float* convB    = (const float*)d_in[5];
    const float* dtW      = (const float*)d_in[6];
    const float* dtB      = (const float*)d_in[7];
    const float* BW       = (const float*)d_in[8];
    const float* CW       = (const float*)d_in[9];
    const float* A_log    = (const float*)d_in[10];
    const float* outW     = (const float*)d_in[11];
    const float* gateW    = (const float*)d_in[12];
    const float* upW      = (const float*)d_in[13];
    const float* downW    = (const float*)d_in[14];
    float* out = (float*)d_out;

    // ---- workspace layout ----
    char* ws = (char*)d_ws;
    size_t off = 0;
    auto alloc = [&](size_t bytes) { char* p = ws + off; off = align_up(off + bytes); return p; };

    unsigned* wp_inproj = (unsigned*)alloc((size_t)D * NP * 2);
    unsigned* wp_dt     = (unsigned*)alloc((size_t)E * E * 2);
    unsigned* wp_B      = (unsigned*)alloc((size_t)E * NS * 2);
    unsigned* wp_C      = (unsigned*)alloc((size_t)E * NS * 2);
    unsigned* wp_out    = (unsigned*)alloc((size_t)E * D * 2);
    unsigned* wp_gate   = (unsigned*)alloc((size_t)D * I * 2);
    unsigned* wp_up     = (unsigned*)alloc((size_t)D * I * 2);
    unsigned* wp_down   = (unsigned*)alloc((size_t)I * D * 2);
    half_t* n1_16      = (half_t*)alloc((size_t)T * D * 2);
    float*  xp32       = (float*) alloc((size_t)T * NP * 4);
    float*  xc32       = (float*) alloc((size_t)T * E * 4);
    half_t* xc16       = (half_t*)alloc((size_t)T * E * 2);
    float*  dt32       = (float*) alloc((size_t)T * E * 4);
    float*  Bp32       = (float*) alloc((size_t)T * NS * 4);
    float*  Cp32       = (float*) alloc((size_t)T * NS * 4);
    float*  y32        = (float*) alloc((size_t)T * E * 4);
    half_t* y16        = (half_t*)alloc((size_t)T * E * 2);
    float*  x2_32      = (float*) alloc((size_t)T * D * 4);
    half_t* n2_16      = (half_t*)alloc((size_t)T * D * 2);
    float*  g32        = (float*) alloc((size_t)T * I * 4);
    float*  u32        = (float*) alloc((size_t)T * I * 4);
    half_t* a16        = (half_t*)alloc((size_t)T * I * 2);

    auto pack = [&](const float* src, unsigned* dst, int K, int N) {
        int n = (K / 2) * N;
        pack_weight_f16<<<(n + 255) / 256, 256, 0, stream>>>(src, dst, K, N);
    };

    // weight conversions into fragment-ready packed layout
    pack(in_projW, wp_inproj, D, NP);
    pack(dtW,      wp_dt,     E, E);
    pack(BW,       wp_B,      E, NS);
    pack(CW,       wp_C,      E, NS);
    pack(outW,     wp_out,    E, D);
    pack(gateW,    wp_gate,   D, I);
    pack(upW,      wp_up,     D, I);
    pack(downW,    wp_down,   I, D);

    // 1) rmsnorm(x) -> f16
    rmsnorm_f16<<<T, 256, 0, stream>>>(x, in_nw, n1_16, D);

    // 2) xp = n1 @ in_proj  (2048 x 4096 x 1024)
    gemm_wmma_f16<0, 128><<<dim3(NP / 128, T / 64), 128, 0, stream>>>(
        n1_16, wp_inproj, xp32, nullptr, T, NP, D);

    // 3) depthwise conv + silu
    conv_silu<<<(T * E + 255) / 256, 256, 0, stream>>>(xp32, convW, convB, xc32, xc16, T, E, NP);

    // 4) dt_raw = xc @ dt_W ; softplus+clip
    gemm_wmma_f16<0, 128><<<dim3(E / 128, T / 64), 128, 0, stream>>>(
        xc16, wp_dt, dt32, nullptr, T, E, E);
    softplus_clip<<<(T * E + 255) / 256, 256, 0, stream>>>(dt32, dtB, T * E, E);

    // 5) Bp / Cp projections (N = 64)
    gemm_wmma_f16<0, 64><<<dim3(NS / 64, T / 64), 128, 0, stream>>>(
        xc16, wp_B, Bp32, nullptr, T, NS, E);
    gemm_wmma_f16<0, 64><<<dim3(NS / 64, T / 64), 128, 0, stream>>>(
        xc16, wp_C, Cp32, nullptr, T, NS, E);

    // 6) sequential selective scan
    ssm_scan<<<E / 64, 64, 0, stream>>>(xc32, dt32, Bp32, Cp32, A_log, y32, T, E);

    // 7) y *= silu(z)
    gate_mul<<<(T * E + 255) / 256, 256, 0, stream>>>(xp32, y32, y16, T, E, NP);

    // 8) x2 = x + y @ out_proj
    gemm_wmma_f16<1, 128><<<dim3(D / 128, T / 64), 128, 0, stream>>>(
        y16, wp_out, x2_32, x, T, D, E);

    // 9) rmsnorm(x2) -> f16
    rmsnorm_f16<<<T, 256, 0, stream>>>(x2_32, ffn_nw, n2_16, D);

    // 10) SwiGLU FFN
    gemm_wmma_f16<0, 128><<<dim3(I / 128, T / 64), 128, 0, stream>>>(
        n2_16, wp_gate, g32, nullptr, T, I, D);
    gemm_wmma_f16<0, 128><<<dim3(I / 128, T / 64), 128, 0, stream>>>(
        n2_16, wp_up, u32, nullptr, T, I, D);
    silu_mul<<<(T * I + 255) / 256, 256, 0, stream>>>(g32, u32, a16, T * I);

    // 11) out = x2 + a @ down_W
    gemm_wmma_f16<1, 128><<<dim3(D / 128, T / 64), 128, 0, stream>>>(
        a16, wp_down, out, x2_32, T, D, I);
}
